// FullQKAttention_8366596293017
// MI455X (gfx1250) — matrix operand order repaired and checked by reading, not compiled, not run on hardware
//
#include <hip/hip_runtime.h>
#include <hip/hip_bf16.h>

typedef __attribute__((ext_vector_type(16))) _Float16 v16h;
typedef __attribute__((ext_vector_type(8)))  _Float16 v8h;
typedef __attribute__((ext_vector_type(4)))  _Float16 v4h;
typedef __attribute__((ext_vector_type(8)))  float    v8f;

#define NB 8
#define SEQ 4096
#define DIM 128
#define SCALE 0.08838834764831845f                  /* 128^-0.5              */
#define K2LOG 0.12753041652967902f                  /* SCALE * log2(e)       */

union AOp { v16h v; v8h h[2]; };

__device__ __forceinline__ float fast_exp2(float x) {
  return __builtin_amdgcn_exp2f(x);                 // v_exp_f32
}

// ---------------------------------------------------------------------------
// Prep 1: per-row L2 norm; Q (raw) and K (normalized) -> f16; row norm saved
// (it is an exact softmax shift bound: |q.k_hat| <= ||q||). One wave per row.
// ---------------------------------------------------------------------------
__global__ __launch_bounds__(256) void prep_qk_kernel(const float* __restrict__ qk,
                                                      _Float16* __restrict__ Qf,
                                                      _Float16* __restrict__ Kf,
                                                      float* __restrict__ Nrm) {
  const int lane = threadIdx.x & 31;
  const int wv   = threadIdx.x >> 5;
  const size_t row = (size_t)blockIdx.x * 8 + wv;          // NB*SEQ rows total
  const float* src = qk + row * DIM + lane * 4;
  float4 x = *(const float4*)src;
  float ss = x.x * x.x + x.y * x.y + x.z * x.z + x.w * x.w;
#pragma unroll
  for (int m = 16; m >= 1; m >>= 1) ss += __shfl_xor(ss, m, 32);
  const float nrm = fmaxf(sqrtf(ss), 1e-12f);
  const float inv = 1.0f / nrm;
  if (lane == 0) Nrm[row] = nrm;
  v4h q, k;
  q[0] = (_Float16)x.x;         q[1] = (_Float16)x.y;
  q[2] = (_Float16)x.z;         q[3] = (_Float16)x.w;
  k[0] = (_Float16)(x.x * inv); k[1] = (_Float16)(x.y * inv);
  k[2] = (_Float16)(x.z * inv); k[3] = (_Float16)(x.w * inv);
  *(v4h*)(Qf + row * DIM + lane * 4) = q;
  *(v4h*)(Kf + row * DIM + lane * 4) = k;
}

// ---------------------------------------------------------------------------
// Prep 2: V (b,s,d) fp32 -> Vt (b,d,s) f16 via LDS 32x32 tile transpose.
// ---------------------------------------------------------------------------
__global__ __launch_bounds__(256) void prep_vt_kernel(const float* __restrict__ v,
                                                      _Float16* __restrict__ Vt) {
  __shared__ float t[32][33];
  const int tx = threadIdx.x & 31;
  const int ty = threadIdx.x >> 5;                 // 0..7
  const int bid = blockIdx.x;
  const int dt = bid & 3;
  const int st = (bid >> 2) & 127;
  const int b  = bid >> 9;
#pragma unroll
  for (int i = 0; i < 4; ++i) {
    const int s = st * 32 + ty + i * 8;
    t[ty + i * 8][tx] = v[((size_t)b * SEQ + s) * DIM + dt * 32 + tx];
  }
  __syncthreads();
#pragma unroll
  for (int i = 0; i < 4; ++i) {
    const int d = dt * 32 + ty + i * 8;
    Vt[((size_t)b * DIM + d) * (size_t)SEQ + st * 32 + tx] = (_Float16)t[tx][ty + i * 8];
  }
}

// ---------------------------------------------------------------------------
// Attention: one wave per 16-row strip; fixed-shift softmax (bound = ||q||/sqrt(d)),
// masks only touched on the diagonal tile, paired independent WMMA chains.
// ---------------------------------------------------------------------------
__global__ __launch_bounds__(256) void attn_kernel(const _Float16* __restrict__ Qf,
                                                   const _Float16* __restrict__ Kf,
                                                   const _Float16* __restrict__ Vt,
                                                   const float* __restrict__ Nrm,
                                                   float* __restrict__ outp,
                                                   float* __restrict__ attnp) {
  __shared__ _Float16 lds[8][2][16][40];   // 80B row stride: 16B aligned, bank-spread

  const int lane  = threadIdx.x & 31;
  const int wave  = threadIdx.x >> 5;
  const int batch = wave;
  const int ipos  = (gridDim.x - 1) - blockIdx.x;      // heavy strips first
  const int ibase = ipos * 16;
  const int njt   = ipos + 1;
  const int hi    = lane >> 4;
  const int lan   = lane & 15;
  const int c0    = hi ? 8 : 0;
  const int koff  = hi ? 16 : 0;

  // ---- preload A operand (Q strip) in contiguous k order (k = 16*hi + s) ----
  AOp a[4];
  const _Float16* qrow = Qf + ((size_t)batch * SEQ + ibase + lan) * DIM;
#pragma unroll
  for (int ks = 0; ks < 4; ++ks) {
    a[ks].h[0] = *(const v8h*)(qrow + ks * 32 + koff);
    a[ks].h[1] = *(const v8h*)(qrow + ks * 32 + koff + 8);
  }

  // per-row exp2 shift: t = c*K2LOG + nb2,  nb2 = -||q||*K2LOG
  float nb2[8];
#pragma unroll
  for (int r = 0; r < 8; ++r)
    nb2[r] = -Nrm[(size_t)batch * SEQ + ibase + r + 8 * hi] * K2LOG;

  const _Float16* kbase = Kf + (size_t)batch * SEQ * DIM;

  auto qk_tile = [&](int jt) -> v8f {
    const _Float16* krow = kbase + (size_t)(jt * 16 + lan) * DIM + koff;
    v8f c = {};
#pragma unroll
    for (int ks = 0; ks < 4; ++ks)
      c = __builtin_amdgcn_wmma_f32_16x16x32_f16(false, a[ks].v, false,
                                                 *(const v16h*)(krow + ks * 32),
                                                 (short)0, c, false, false);
    return c;
  };
  auto qk_tile2 = [&](int jt, v8f& ca, v8f& cb) {   // two independent chains
    const _Float16* k0 = kbase + (size_t)(jt * 16 + lan) * DIM + koff;
    const _Float16* k1 = k0 + 16 * DIM;
    __builtin_prefetch(k0 + 32 * DIM, 0, 1);
    ca = (v8f){}; cb = (v8f){};
#pragma unroll
    for (int ks = 0; ks < 4; ++ks) {
      ca = __builtin_amdgcn_wmma_f32_16x16x32_f16(false, a[ks].v, false,
                                                  *(const v16h*)(k0 + ks * 32),
                                                  (short)0, ca, false, false);
      cb = __builtin_amdgcn_wmma_f32_16x16x32_f16(false, a[ks].v, false,
                                                  *(const v16h*)(k1 + ks * 32),
                                                  (short)0, cb, false, false);
    }
  };

  // ------------------------------ pass 1: l ------------------------------
  float l[8];
#pragma unroll
  for (int r = 0; r < 8; ++r) l[r] = 0.0f;

  auto acc_l = [&](const v8f& c) {
#pragma unroll
    for (int r = 0; r < 8; ++r)
      l[r] += fast_exp2(fmaf(c[r], K2LOG, nb2[r]));
  };

  {
    const int nfull = njt - 1;                       // tiles strictly below diag
    int jt = 0;
    for (; jt + 1 < nfull; jt += 2) { v8f ca, cb; qk_tile2(jt, ca, cb); acc_l(ca); acc_l(cb); }
    if (jt < nfull) { v8f c = qk_tile(jt); acc_l(c); }
    // diagonal tile: exclude diag (-50000 -> exp == 0 exactly) and causal cols
    v8f c = qk_tile(njt - 1);
    const int col = (njt - 1) * 16 + lan;
#pragma unroll
    for (int r = 0; r < 8; ++r) {
      const int row = ibase + r + 8 * hi;
      const float e = fast_exp2(fmaf(c[r], K2LOG, nb2[r]));
      l[r] += (col >= row) ? 0.0f : e;
    }
  }
  // sum across 16-lane half-wave
#pragma unroll
  for (int r = 0; r < 8; ++r) {
#pragma unroll
    for (int dd = 1; dd < 16; dd <<= 1) l[r] += __shfl_xor(l[r], dd, 16);
  }
  float rl[8], pd[8];
#pragma unroll
  for (int r = 0; r < 8; ++r) {
    rl[r] = (l[r] > 0.0f) ? 1.0f / l[r] : 0.0f;      // l==0 only for global row 0
    pd[r] = (l[r] > 0.0f) ? 0.0f : 1.0f;             // diagonal prob fixup
  }

  // --------------------- pass 2: attn out + PV accumulate ---------------------
  v8f o[8];
#pragma unroll
  for (int db = 0; db < 8; ++db) o[db] = (v8f){};

  const _Float16* vtb = Vt + (size_t)batch * DIM * SEQ;

  auto emit = [&](const v8f& c, int jt, int buf, int hf) {
    const int col = jt * 16 + lan;
    const bool isdiag = (jt == njt - 1);             // wave-uniform
#pragma unroll
    for (int r = 0; r < 8; ++r) {
      const int row = ibase + r + 8 * hi;
      float p = fast_exp2(fmaf(c[r], K2LOG, nb2[r])) * rl[r];
      if (isdiag) {
        p = (col == row) ? pd[r] : p;
        p = (col >  row) ? 0.0f  : p;
      }
      attnp[((size_t)batch * SEQ + row) * (size_t)SEQ + col] = p;
      lds[wave][buf][r + 8 * hi][hf * 16 + lan] = (_Float16)p;
    }
  };

  for (int jc = 0; jc < njt; jc += 2) {
    const int buf = (jc >> 1) & 1;
    if (jc + 1 < njt) {
      v8f ca, cb;
      qk_tile2(jc, ca, cb);
      emit(ca, jc, buf, 0);
      emit(cb, jc + 1, buf, 1);
    } else {
      v8f ca = qk_tile(jc);
      emit(ca, jc, buf, 0);
#pragma unroll
      for (int r = 0; r < 8; ++r)
        lds[wave][buf][r + 8 * hi][16 + lan] = (_Float16)0.0f;
    }
    // wave-private LDS handoff (DS in-order per wave); drain + compiler fence
    asm volatile("s_wait_dscnt 0x0" ::: "memory");
    AOp pa;
    const _Float16* lrow = &lds[wave][buf][lan][0];
    pa.h[0] = *(const v8h*)(lrow + koff);
    pa.h[1] = *(const v8h*)(lrow + koff + 8);
#pragma unroll
    for (int db = 0; db < 8; ++db) {                  // 8 independent chains
      const _Float16* vrow = vtb + (size_t)(db * 16 + lan) * SEQ + jc * 16 + koff;
      o[db] = __builtin_amdgcn_wmma_f32_16x16x32_f16(false, pa.v, false,
                                                     *(const v16h*)vrow,
                                                     (short)0, o[db], false, false);
    }
  }

  // ------------------------------- store out ---------------------------------
#pragma unroll
  for (int db = 0; db < 8; ++db) {
#pragma unroll
    for (int r = 0; r < 8; ++r) {
      outp[((size_t)batch * SEQ + ibase + r + 8 * hi) * (size_t)DIM + db * 16 + lan] =
          o[db][r];
    }
  }

  // -------------------- zero-fill masked region of attn ----------------------
  const float4 z = make_float4(0.0f, 0.0f, 0.0f, 0.0f);
  for (int r = 0; r < 16; ++r) {
    float* rowp = attnp + ((size_t)batch * SEQ + ibase + r) * (size_t)SEQ;
    for (int col = njt * 16 + lane * 4; col < SEQ; col += 128) {
      *(float4*)(rowp + col) = z;
    }
  }
}

// ---------------------------------------------------------------------------
extern "C" void kernel_launch(void* const* d_in, const int* in_sizes, int n_in,
                              void* d_out, int out_size, void* d_ws, size_t ws_size,
                              hipStream_t stream) {
  (void)in_sizes; (void)n_in; (void)out_size; (void)ws_size;
  const float* qk = (const float*)d_in[0];
  const float* v  = (const float*)d_in[1];
  float* outp  = (float*)d_out;
  float* attnp = outp + (size_t)NB * SEQ * DIM;

  _Float16* Qf = (_Float16*)d_ws;                      // 8 MB
  _Float16* Kf = Qf + (size_t)NB * SEQ * DIM;          // 8 MB
  _Float16* Vt = Kf + (size_t)NB * SEQ * DIM;          // 8 MB
  float*    Nrm = (float*)(Vt + (size_t)NB * SEQ * DIM); // 128 KB

  prep_qk_kernel<<<NB * SEQ / 8, 256, 0, stream>>>(qk, Qf, Kf, Nrm);
  prep_vt_kernel<<<NB * (SEQ / 32) * (DIM / 32), 256, 0, stream>>>(v, Vt);
  attn_kernel<<<SEQ / 16, 256, 0, stream>>>(Qf, Kf, Vt, Nrm, outp, attnp);
}
